// Encoder_45475113730366
// MI455X (gfx1250) — compile-verified, matrix operand order checked by reading, not run on hardware
//
#include <hip/hip_runtime.h>
#include <hip/hip_bf16.h>

#define B_    64
#define S_    2048
#define DM_   128
#define H_    8
#define DK_   16
#define DI_   512
#define L_    4
#define NROWS (B_*S_)          /* 131072 */
#define EPS_  1e-6f
#define TEMP_ 2.0f             /* 0.5*sqrt(16) */

typedef __attribute__((ext_vector_type(16))) _Float16 v16h;
typedef __attribute__((ext_vector_type(8)))  float    v8f;

// ---------------- WMMA fragment helpers (CDNA5 wave32, 16x16x32 f16) --------
// A (16x32 f16): lane m = lane&15; lanes<16 hold K 0..7 / 16..23, lanes>=16 hold
// K 8..15 / 24..31 (ISA 7.12.2).
__device__ inline v16h load_a_frag(const _Float16* tile, int lane, int kslab, int ld) {
  const int m  = lane & 15;
  const int kb = (lane >> 4) << 3;
  const _Float16* p = tile + m * ld + kslab * 32 + kb;
  v16h a;
#pragma unroll
  for (int i = 0; i < 8; ++i) { a[i] = p[i]; a[8 + i] = p[16 + i]; }
  return a;
}

// B fragments pre-packed by repack_w: contiguous 16 halves per lane.
__device__ inline v16h load_b_frag(const _Float16* packed, int lane, int kslab,
                                   int tn, int numTiles) {
  const _Float16* p = packed + ((size_t)(kslab * numTiles + tn) * 32 + lane) * 16;
  v16h b;
#pragma unroll
  for (int i = 0; i < 16; ++i) b[i] = p[i];
  return b;
}

__device__ inline v8f wmma_f16(v16h a, v16h b, v8f c) {
  return __builtin_amdgcn_wmma_f32_16x16x32_f16(false, a, false, b, (short)0, c,
                                                false, false);
}

// ---------------- weight repack: f32 row-major [K][N] -> f16 B fragments ----
__global__ __launch_bounds__(256) void repack_w(const float* __restrict__ W,
                                                _Float16* __restrict__ out,
                                                int K, int N) {
  const int layer = blockIdx.y;
  const size_t nelem = (size_t)K * N;
  const size_t tid = (size_t)blockIdx.x * blockDim.x + threadIdx.x;
  if (tid >= nelem) return;
  const int i    = (int)(tid & 15);
  const int lane = (int)((tid >> 4) & 31);
  const int rest = (int)(tid >> 9);
  const int numTiles = N >> 4;
  const int tn   = rest % numTiles;
  const int slab = rest / numTiles;
  const int k = slab * 32 + ((i >> 3) << 4) + ((lane >> 4) << 3) + (i & 7);
  const int n = tn * 16 + (lane & 15);
  out[layer * nelem + tid] = (_Float16)W[layer * nelem + (size_t)k * N + n];
}

// ---------------- kernel 1: LN1 + q/k/v projections + k,v pair-halving ------
__global__ __launch_bounds__(256) void qkv_kernel(
    const float* __restrict__ Xq, int ldq,
    const float* __restrict__ Xk, int ldk,
    const float* __restrict__ Xv, int ldv,
    const float* __restrict__ g1, const float* __restrict__ b1ln,
    const _Float16* __restrict__ pWq, const _Float16* __restrict__ pWk,
    const _Float16* __restrict__ pWv,
    float* __restrict__ qout, float* __restrict__ khout, float* __restrict__ vhout)
{
  __shared__ _Float16 aQ[16][136];
  __shared__ _Float16 aK[16][136];
  __shared__ _Float16 aV[16][136];
  __shared__ float red[16][16];
  __shared__ float red2[16][16];
  __shared__ float dT[8][16][16];

  const int tid  = threadIdx.x;
  const int row0 = blockIdx.x * 16;
  const int r = tid >> 4;
  const int c = tid & 15;

  float x[8];
  float s = 0.f, sq = 0.f;
  const float* xr = Xq + (size_t)(row0 + r) * ldq + c * 8;
#pragma unroll
  for (int j = 0; j < 8; ++j) { x[j] = xr[j]; s += x[j]; sq += x[j] * x[j]; }
  red[r][c] = s; red2[r][c] = sq;
  const float* kr = Xk + (size_t)(row0 + r) * ldk + c * 8;
  const float* vr = Xv + (size_t)(row0 + r) * ldv + c * 8;
#pragma unroll
  for (int j = 0; j < 8; ++j) {
    aK[r][c * 8 + j] = (_Float16)kr[j];
    aV[r][c * 8 + j] = (_Float16)vr[j];
  }
  __syncthreads();
  if (c == 0) {
    float ts = 0.f, tq = 0.f;
    for (int i = 0; i < 16; ++i) { ts += red[r][i]; tq += red2[r][i]; }
    const float mu  = ts * (1.f / 128.f);
    const float var = tq * (1.f / 128.f) - mu * mu;
    red[r][0]  = mu;
    red2[r][0] = rsqrtf(var + EPS_);
  }
  __syncthreads();
  {
    const float mu = red[r][0], ri = red2[r][0];
#pragma unroll
    for (int j = 0; j < 8; ++j) {
      const int col = c * 8 + j;
      aQ[r][col] = (_Float16)(((x[j] - mu) * ri) * g1[col] + b1ln[col]);
    }
  }
  __syncthreads();

  const int w = tid >> 5;            // wave == head
  const int lane = tid & 31;
  const int n = lane & 15;
  const int mb = (lane >> 4) << 3;

  { // q head tile
    v8f acc = {};
#pragma unroll
    for (int ks = 0; ks < 4; ++ks)
      acc = wmma_f16(load_a_frag(&aQ[0][0], lane, ks, 136),
                     load_b_frag(pWq, lane, ks, w, 8), acc);
#pragma unroll
    for (int j = 0; j < 8; ++j)
      qout[(size_t)(row0 + mb + j) * 128 + w * 16 + n] = acc[j];
  }
  { // k head tile -> pair mean
    v8f acc = {};
#pragma unroll
    for (int ks = 0; ks < 4; ++ks)
      acc = wmma_f16(load_a_frag(&aK[0][0], lane, ks, 136),
                     load_b_frag(pWk, lane, ks, w, 8), acc);
#pragma unroll
    for (int j = 0; j < 8; ++j) dT[w][mb + j][n] = acc[j];
    __syncthreads();
#pragma unroll
    for (int j = 0; j < 4; ++j) {
      const int t = lane * 4 + j;
      const int m = t >> 3, e = t & 7;
      khout[(size_t)(row0 + m) * 64 + w * 8 + e] =
          0.5f * (dT[w][m][2 * e] + dT[w][m][2 * e + 1]);
    }
    __syncthreads();
  }
  { // v head tile -> pair mean
    v8f acc = {};
#pragma unroll
    for (int ks = 0; ks < 4; ++ks)
      acc = wmma_f16(load_a_frag(&aV[0][0], lane, ks, 136),
                     load_b_frag(pWv, lane, ks, w, 8), acc);
#pragma unroll
    for (int j = 0; j < 8; ++j) dT[w][mb + j][n] = acc[j];
    __syncthreads();
#pragma unroll
    for (int j = 0; j < 4; ++j) {
      const int t = lane * 4 + j;
      const int m = t >> 3, e = t & 7;
      vhout[(size_t)(row0 + m) * 64 + w * 8 + e] =
          0.5f * (dT[w][m][2 * e] + dT[w][m][2 * e + 1]);
    }
  }
}

// ---------------- kernel 2: attn[b,h,16,8] = softmax(q^T kh / TEMP) ---------
__global__ __launch_bounds__(256) void attn_kernel(
    const float* __restrict__ qbuf, const float* __restrict__ khbuf,
    const unsigned char* __restrict__ mask, float* __restrict__ attnOut)
{
  const int b = blockIdx.x >> 3;
  const int h = blockIdx.x & 7;
  const int t = threadIdx.x;
  const int de = t & 127, d = de >> 3, e = de & 7, part = t >> 7;
  const float* qp = qbuf  + (size_t)b * S_ * 128 + h * 16 + d;
  const float* kp = khbuf + (size_t)b * S_ * 64  + h * 8  + e;
  float s = 0.f;
  for (int l = part; l < S_; l += 2)
    s += qp[(size_t)l * 128] * kp[(size_t)l * 64];
  __shared__ float acc[2][128];
  __shared__ float sm[16][8];
  acc[part][de] = s;
  __syncthreads();
  if (t < 128) {
    float v = (acc[0][de] + acc[1][de]) * (1.f / TEMP_);
    if (!mask[(size_t)b * 128 + de]) v = -1e9f;
    sm[d][e] = v;
  }
  __syncthreads();
  if (t < 16) {
    float mx = -1e30f;
    for (int j = 0; j < 8; ++j) mx = fmaxf(mx, sm[t][j]);
    float ssum = 0.f, ex[8];
    for (int j = 0; j < 8; ++j) { ex[j] = __expf(sm[t][j] - mx); ssum += ex[j]; }
    const float inv = 1.f / ssum;
    for (int j = 0; j < 8; ++j)
      attnOut[((size_t)(b * 8 + h) * 16 + t) * 8 + j] = ex[j] * inv;
  }
}

// ---------------- kernel 3: attn-apply + Wfc + res + LN2 + mish-FFN + res ---
__global__ __launch_bounds__(256) void out_ffn_kernel(
    const float* __restrict__ Xres, int ldres,
    const float* __restrict__ vhbuf, const float* __restrict__ attnbuf,
    const _Float16* __restrict__ pWfc,
    const float* __restrict__ g2, const float* __restrict__ b2ln,
    const _Float16* __restrict__ pW1, const float* __restrict__ bias1,
    const _Float16* __restrict__ pW2, const float* __restrict__ bias2,
    float* __restrict__ Y)
{
  __shared__ float    vhT[1024];
  __shared__ float    attnS[1024];
  __shared__ _Float16 aoT[16][136];
  __shared__ float    midT[16][128];
  __shared__ _Float16 xnT[16][136];
  __shared__ _Float16 hT[16][520];
  __shared__ float    red[16][16], red2[16][16];

  const int tid  = threadIdx.x;
  const int row0 = blockIdx.x * 16;
  const int bidx = row0 >> 11;

#pragma unroll
  for (int j = 0; j < 4; ++j) {
    const int p = tid * 4 + j;
    vhT[p]   = vhbuf[(size_t)row0 * 64 + p];
    attnS[p] = attnbuf[(size_t)bidx * 1024 + p];
  }
  __syncthreads();
#pragma unroll
  for (int j = 0; j < 8; ++j) {  // ao[l][d*H+h] = sum_e attn[h,d,e]*vh[l,h,e]
    const int p = tid * 8 + j;
    const int r = p >> 7, col = p & 127;
    const int d = col >> 3, h = col & 7;
    float s = 0.f;
#pragma unroll
    for (int e = 0; e < 8; ++e) s += attnS[h * 128 + d * 8 + e] * vhT[r * 64 + h * 8 + e];
    aoT[r][col] = (_Float16)s;
  }
  __syncthreads();

  const int w = tid >> 5, lane = tid & 31;
  const int n = lane & 15, mb = (lane >> 4) << 3;

  { // mid = ao @ Wfc + residual
    v8f acc = {};
#pragma unroll
    for (int ks = 0; ks < 4; ++ks)
      acc = wmma_f16(load_a_frag(&aoT[0][0], lane, ks, 136),
                     load_b_frag(pWfc, lane, ks, w, 8), acc);
#pragma unroll
    for (int j = 0; j < 8; ++j) {
      const int m = mb + j;
      midT[m][w * 16 + n] = acc[j] + Xres[(size_t)(row0 + m) * ldres + w * 16 + n];
    }
  }
  __syncthreads();
  { // LN2
    const int r = tid >> 4, c = tid & 15;
    float x[8], s = 0.f, sq = 0.f;
#pragma unroll
    for (int j = 0; j < 8; ++j) { x[j] = midT[r][c * 8 + j]; s += x[j]; sq += x[j] * x[j]; }
    red[r][c] = s; red2[r][c] = sq;
    __syncthreads();
    if (c == 0) {
      float ts = 0.f, tq = 0.f;
      for (int i = 0; i < 16; ++i) { ts += red[r][i]; tq += red2[r][i]; }
      const float mu = ts * (1.f / 128.f), var = tq * (1.f / 128.f) - mu * mu;
      red[r][0] = mu; red2[r][0] = rsqrtf(var + EPS_);
    }
    __syncthreads();
    const float mu = red[r][0], ri = red2[r][0];
#pragma unroll
    for (int j = 0; j < 8; ++j) {
      const int col = c * 8 + j;
      xnT[r][col] = (_Float16)(((x[j] - mu) * ri) * g2[col] + b2ln[col]);
    }
  }
  __syncthreads();
#pragma unroll
  for (int it = 0; it < 4; ++it) { // h = mish(xn @ W1 + b1), 32 tiles / 8 waves
    const int tn = w + it * 8;
    v8f acc = {};
#pragma unroll
    for (int ks = 0; ks < 4; ++ks)
      acc = wmma_f16(load_a_frag(&xnT[0][0], lane, ks, 136),
                     load_b_frag(pW1, lane, ks, tn, 32), acc);
#pragma unroll
    for (int j = 0; j < 8; ++j) {
      const int m = mb + j;
      const float xv = acc[j] + bias1[tn * 16 + n];
      const float sp = (xv > 20.f) ? xv : log1pf(__expf(xv));
      hT[m][tn * 16 + n] = (_Float16)(xv * tanhf(sp));
    }
  }
  __syncthreads();
  { // y = h @ W2 + b2 + mid
    v8f acc = {};
#pragma unroll
    for (int ks = 0; ks < 16; ++ks)
      acc = wmma_f16(load_a_frag(&hT[0][0], lane, ks, 520),
                     load_b_frag(pW2, lane, ks, w, 8), acc);
#pragma unroll
    for (int j = 0; j < 8; ++j) {
      const int m = mb + j, col = w * 16 + n;
      Y[(size_t)(row0 + m) * 128 + col] = acc[j] + bias2[col] + midT[m][col];
    }
  }
}

// ---------------- kernel 4: enc = concat(e1,e2) @ WL2 + bL2 ----------------
__global__ __launch_bounds__(256) void wl2_kernel(
    const float* __restrict__ e1, const float* __restrict__ e2,
    const _Float16* __restrict__ pWL2, const float* __restrict__ bL2,
    float* __restrict__ enc)
{
  __shared__ _Float16 aT[16][264];
  const int tid = threadIdx.x;
  const int row0 = blockIdx.x * 16;
  const int r = tid >> 4, c = tid & 15;
#pragma unroll
  for (int j = 0; j < 8; ++j) {
    aT[r][c * 8 + j]       = (_Float16)e1[(size_t)(row0 + r) * 128 + c * 8 + j];
    aT[r][128 + c * 8 + j] = (_Float16)e2[(size_t)(row0 + r) * 128 + c * 8 + j];
  }
  __syncthreads();
  const int w = tid >> 5, lane = tid & 31;
  const int n = lane & 15, mb = (lane >> 4) << 3;
  v8f acc = {};
#pragma unroll
  for (int ks = 0; ks < 8; ++ks)
    acc = wmma_f16(load_a_frag(&aT[0][0], lane, ks, 264),
                   load_b_frag(pWL2, lane, ks, w, 8), acc);
#pragma unroll
  for (int j = 0; j < 8; ++j) {
    const int col = w * 16 + n;
    enc[(size_t)(row0 + mb + j) * 128 + col] = acc[j] + bL2[col];
  }
}

extern "C" void kernel_launch(void* const* d_in, const int* in_sizes, int n_in,
                              void* d_out, int out_size, void* d_ws, size_t ws_size,
                              hipStream_t stream) {
  (void)in_sizes; (void)n_in; (void)out_size; (void)ws_size;
  const float* src  = (const float*)d_in[0];
  const unsigned char* mask = (const unsigned char*)d_in[1];
  const float* ln1g = (const float*)d_in[2];
  const float* ln1b = (const float*)d_in[3];
  const float* Wq   = (const float*)d_in[4];
  const float* Wk   = (const float*)d_in[5];
  const float* Wv   = (const float*)d_in[6];
  const float* Wfc  = (const float*)d_in[7];
  const float* ln2g = (const float*)d_in[8];
  const float* ln2b = (const float*)d_in[9];
  const float* W1   = (const float*)d_in[10];
  const float* b1   = (const float*)d_in[11];
  const float* W2   = (const float*)d_in[12];
  const float* b2   = (const float*)d_in[13];
  const float* WL2  = (const float*)d_in[14];
  const float* bL2  = (const float*)d_in[15];

  float* enc = (float*)d_out;
  float* m1  = enc + (size_t)NROWS * 128;
  float* m2  = m1  + (size_t)NROWS * 128;

  float* qbuf  = (float*)d_ws;
  float* khbuf = qbuf  + (size_t)NROWS * 128;
  float* vhbuf = khbuf + (size_t)NROWS * 64;
  float* attnb = vhbuf + (size_t)NROWS * 64;
  float* e2buf = attnb + (size_t)B_ * H_ * 16 * 8;
  _Float16* wp   = (_Float16*)(e2buf + (size_t)NROWS * 128);
  _Float16* pWq  = wp;
  _Float16* pWk  = pWq  + 4 * 16384;
  _Float16* pWv  = pWk  + 4 * 16384;
  _Float16* pWfc = pWv  + 4 * 16384;
  _Float16* pW1  = pWfc + 4 * 16384;
  _Float16* pW2  = pW1  + 4 * 65536;
  _Float16* pWL2 = pW2  + 4 * 65536;
  float* e1buf = enc;  // reuse: enc slot is only produced after e-stacks finish

  repack_w<<<dim3(64, 4),  256, 0, stream>>>(Wq,  pWq,  128, 128);
  repack_w<<<dim3(64, 4),  256, 0, stream>>>(Wk,  pWk,  128, 128);
  repack_w<<<dim3(64, 4),  256, 0, stream>>>(Wv,  pWv,  128, 128);
  repack_w<<<dim3(64, 4),  256, 0, stream>>>(Wfc, pWfc, 128, 128);
  repack_w<<<dim3(256, 4), 256, 0, stream>>>(W1,  pW1,  128, 512);
  repack_w<<<dim3(256, 4), 256, 0, stream>>>(W2,  pW2,  512, 128);
  repack_w<<<dim3(128, 1), 256, 0, stream>>>(WL2, pWL2, 256, 128);

  const int NB = NROWS / 16;
  auto layer = [&](int i, const float* q, int ldq, const float* k, int ldk,
                   const float* v, int ldv, float* out) {
    qkv_kernel<<<NB, 256, 0, stream>>>(q, ldq, k, ldk, v, ldv,
        ln1g + i * 128, ln1b + i * 128,
        pWq + i * 16384, pWk + i * 16384, pWv + i * 16384,
        qbuf, khbuf, vhbuf);
    attn_kernel<<<B_ * H_, 256, 0, stream>>>(qbuf, khbuf, mask, attnb);
    out_ffn_kernel<<<NB, 256, 0, stream>>>(q, ldq, vhbuf, attnb,
        pWfc + i * 16384, ln2g + i * 128, ln2b + i * 128,
        pW1 + i * 65536, b1 + i * 512, pW2 + i * 65536, b2 + i * 128, out);
  };
  const float* lo = src;        // src[:,:,:128], row stride 256
  const float* hi = src + 128;  // src[:,:,128:]

  layer(0, lo, 256, lo, 256, lo, 256, m1);
  for (int i = 1; i < 4; ++i) layer(i, m1, 128, m1, 128, m1, 128, m1);

  layer(0, hi, 256, hi, 256, hi, 256, m2);
  for (int i = 1; i < 4; ++i) layer(i, m2, 128, m2, 128, m2, 128, m2);

  layer(0, hi, 256, lo, 256, lo, 256, e1buf);
  for (int i = 1; i < 4; ++i) layer(i, e1buf, 128, e1buf, 128, e1buf, 128, e1buf);

  layer(0, lo, 256, hi, 256, hi, 256, e2buf);
  for (int i = 1; i < 4; ++i) layer(i, e2buf, 128, e2buf, 128, e2buf, 128, e2buf);

  wl2_kernel<<<NB, 256, 0, stream>>>(e1buf, e2buf, pWL2, bL2, enc);
  for (int i = 0; i < 4; ++i) layer(i, enc, 128, enc, 128, enc, 128, enc);
}
